// sLSTM_57775900066442
// MI455X (gfx1250) — compile-verified, matrix operand order checked by reading, not run on hardware
//
#include <hip/hip_runtime.h>
#include <hip/hip_bf16.h>
#include <math.h>

typedef __attribute__((ext_vector_type(16))) _Float16 v16h;
typedef __attribute__((ext_vector_type(8)))  _Float16 v8h;
typedef __attribute__((ext_vector_type(2)))  _Float16 v2h;
typedef __attribute__((ext_vector_type(8)))  float    v8f;

#define DIMK   1024      // model dim / reduction dim
#define NCOLS  4096      // 4*DIM
#define MROWS  16384     // B*T
#define TSEQ   4096
#define NBATCH 4
#define TILE_M 64
#define TILE_N 64
#define TILE_K 32
#define LDS_STRIDE 40    // 32 + 8 halves pad: 80B rows, v8h loads stay 16B aligned

#define CH 64            // chunks per sequence
#define CL 64            // steps per chunk (CH*CL == TSEQ)

// 16x32 f16 fragment (A-layout, CDNA5 ISA 7.12.2) from LDS tile [16][LDS_STRIDE].
// Used for A (row=M) and for B via the transposed tile (row=N).
__device__ inline v16h load_frag16x32(const _Float16* tile, int lane) {
    const int row = lane & 15;
    const int kb0 = (lane < 16) ? 0 : 8;
    const int kb1 = kb0 + 16;
    v8h lo = *(const v8h*)(tile + row * LDS_STRIDE + kb0);
    v8h hi = *(const v8h*)(tile + row * LDS_STRIDE + kb1);
    v16h r;
#pragma unroll
    for (int i = 0; i < 8; ++i) { r[i] = lo[i]; r[8 + i] = hi[i]; }
    return r;
}

// -------- Kernel A: pre = x @ W + b  (WMMA f16 -> f32 accumulate) --------
__global__ __launch_bounds__(256) void gemm_pre_kernel(
    const float* __restrict__ X,    // [MROWS, DIMK]
    const float* __restrict__ W,    // [DIMK, NCOLS]
    const float* __restrict__ bias, // [NCOLS]
    float* __restrict__ pre)        // [MROWS, NCOLS]
{
    __shared__ __align__(16) _Float16 sA[TILE_M * LDS_STRIDE]; // A tile [m][k]
    __shared__ __align__(16) _Float16 sB[TILE_N * LDS_STRIDE]; // B^T tile [n][k]

    const int tid  = threadIdx.x;
    const int lane = tid & 31;
    const int wave = tid >> 5;   // 0..7
    const int wm   = wave >> 2;  // 0..1
    const int wn   = wave & 3;   // 0..3
    const int m0   = blockIdx.y * TILE_M;
    const int n0   = blockIdx.x * TILE_N;

    v8f acc0 = {};
    v8f acc1 = {};

    for (int kk = 0; kk < DIMK; kk += TILE_K) {
        // Stage A: 64x32, pairs along k: float2 load -> packed 2xf16 -> b32 store
#pragma unroll
        for (int i = 0; i < 4; ++i) {
            int p = tid + i * 256;          // 0..1023 pairs
            int r = p >> 4, c2 = p & 15;    // row, k-pair
            float2 xv = *(const float2*)(X + (size_t)(m0 + r) * DIMK + kk + 2 * c2);
            *(v2h*)(&sA[r * LDS_STRIDE + 2 * c2]) = (v2h){(_Float16)xv.x, (_Float16)xv.y};
        }
        // Stage B^T: [n][k]; two coalesced-in-n loads (k, k+1), packed b32 store
#pragma unroll
        for (int i = 0; i < 4; ++i) {
            int p = tid + i * 256;          // 0..1023 pairs
            int n = p & 63, kp = p >> 6;    // n, k-pair (0..15)
            float w0 = W[(size_t)(kk + 2 * kp)     * NCOLS + n0 + n];
            float w1 = W[(size_t)(kk + 2 * kp + 1) * NCOLS + n0 + n];
            *(v2h*)(&sB[n * LDS_STRIDE + 2 * kp]) = (v2h){(_Float16)w0, (_Float16)w1};
        }
        __syncthreads();

        v16h bf = load_frag16x32(sB + (wn * 16) * LDS_STRIDE, lane);
        v16h a0 = load_frag16x32(sA + (wm * 32) * LDS_STRIDE, lane);
        v16h a1 = load_frag16x32(sA + (wm * 32 + 16) * LDS_STRIDE, lane);

        acc0 = __builtin_amdgcn_wmma_f32_16x16x32_f16(false, a0, false, bf,
                                                      (short)0, acc0, false, false);
        acc1 = __builtin_amdgcn_wmma_f32_16x16x32_f16(false, a1, false, bf,
                                                      (short)0, acc1, false, false);
        __syncthreads();
    }

    // Epilogue (D layout: lane&15 = N; lanes>=16 start at M=8; VGPR i -> row +i)
    const int ncol = n0 + wn * 16 + (lane & 15);
    const float bv = bias[ncol];
    const int mbase = m0 + wm * 32 + ((lane < 16) ? 0 : 8);
#pragma unroll
    for (int i = 0; i < 8; ++i) {
        int m = mbase + i;
        __builtin_nontemporal_store(acc0[i] + bv, &pre[(size_t)m * NCOLS + ncol]);
        __builtin_nontemporal_store(acc1[i] + bv, &pre[(size_t)(m + 16) * NCOLS + ncol]);
    }
}

// Shared gate activations
__device__ inline void gates(const float* __restrict__ p, int d, float ib,
                             float& i_t, float& f_t, float& o_t, float& tz) {
    float pi = __builtin_nontemporal_load(p + d);
    float pf = __builtin_nontemporal_load(p + 1024 + d);
    float po = __builtin_nontemporal_load(p + 2048 + d);
    float pz = __builtin_nontemporal_load(p + 3072 + d);
    i_t = 10.f * tanhf((pi + ib) * 0.1f);                       // softcap
    f_t = -(fmaxf(pf, 0.f) + log1pf(__expf(-fabsf(pf))));       // -softplus
    o_t = 1.f / (1.f + __expf(-po));                            // sigmoid
    tz  = tanhf(pz);
}

// -------- Kernel B1: per-chunk summaries {F, c_loc, n_loc, m_loc} --------
// One thread per (b, chunk, d): B*CH*DIM = 262144 threads.
__global__ __launch_bounds__(256) void scan_chunk_summary_kernel(
    const float* __restrict__ pre, const float* __restrict__ input_bias,
    float* __restrict__ sumF, float* __restrict__ sumC,
    float* __restrict__ sumN, float* __restrict__ sumM)
{
    const size_t g = (size_t)blockIdx.x * 256 + threadIdx.x;
    const int d  = (int)(g & 1023);
    const int bc = (int)(g >> 10);   // b*CH + chunk
    const int b  = bc >> 6;
    const int ch = bc & (CH - 1);
    const float ib = input_bias[d];

    float F = 0.f, c = 0.f, n = 0.f, m = 0.f;
    const float* base = pre + (size_t)(b * TSEQ + ch * CL) * NCOLS;
    for (int t = 0; t < CL; ++t) {
        float i_t, f_t, o_t, tz;
        gates(base + (size_t)t * NCOLS, d, ib, i_t, f_t, o_t, tz);
        F += f_t;
        float mn  = fmaxf(f_t + m, i_t);
        float i_s = __expf(i_t - mn);
        float f_s = __expf(f_t + m - mn);
        c = f_s * c + i_s * tz;
        n = f_s * n + i_s;
        m = mn;
    }
    sumF[g] = F; sumC[g] = c; sumN[g] = n; sumM[g] = m;
}

// -------- Kernel B2: sequential composition over CH chunk maps --------
// One thread per (b, d): records each chunk's INCOMING state.
__global__ __launch_bounds__(256) void scan_compose_kernel(
    const float* __restrict__ sumF, const float* __restrict__ sumC,
    const float* __restrict__ sumN, const float* __restrict__ sumM,
    float* __restrict__ stC, float* __restrict__ stN, float* __restrict__ stM)
{
    const int g = blockIdx.x * 256 + threadIdx.x;   // 0..4095
    const int b = g >> 10;
    const int d = g & 1023;

    float c = 0.f, n = 0.f, m = 0.f;
    for (int k = 0; k < CH; ++k) {
        size_t idx = ((size_t)(b * CH + k) << 10) + d;
        stC[idx] = c; stN[idx] = n; stM[idx] = m;
        float F  = sumF[idx], cl = sumC[idx];
        float nl = sumN[idx], ml = sumM[idx];
        float mn = fmaxf(m + F, ml);
        float ea = __expf(m + F - mn);
        float eb = __expf(ml - mn);
        c = c * ea + cl * eb;
        n = n * ea + nl * eb;
        m = mn;
    }
}

// -------- Kernel B3: replay chunks from incoming state, emit h --------
__global__ __launch_bounds__(256) void scan_replay_kernel(
    const float* __restrict__ pre, const float* __restrict__ input_bias,
    const float* __restrict__ stC, const float* __restrict__ stN,
    const float* __restrict__ stM, float* __restrict__ h_out)
{
    const size_t g = (size_t)blockIdx.x * 256 + threadIdx.x;
    const int d  = (int)(g & 1023);
    const int bc = (int)(g >> 10);
    const int b  = bc >> 6;
    const int ch = bc & (CH - 1);
    const float ib = input_bias[d];

    float c = stC[g], n = stN[g], m = stM[g];
    const int t0 = ch * CL;
    const float* base = pre + (size_t)(b * TSEQ + t0) * NCOLS;
    for (int t = 0; t < CL; ++t) {
        float i_t, f_t, o_t, tz;
        gates(base + (size_t)t * NCOLS, d, ib, i_t, f_t, o_t, tz);
        float mn  = fmaxf(f_t + m, i_t);
        float i_s = __expf(i_t - mn);
        float f_s = __expf(f_t + m - mn);
        c = f_s * c + i_s * tz;
        n = f_s * n + i_s;
        m = mn;
        float h = o_t * (c / (n + 1e-6f));
        __builtin_nontemporal_store(h, &h_out[(size_t)(b * TSEQ + t0 + t) * DIMK + d]);
    }
}

// -------- Kernel C: RMSNorm in place over last dim (1024) --------
__global__ __launch_bounds__(256) void rmsnorm_kernel(
    float* __restrict__ out, const float* __restrict__ scale)
{
    __shared__ float red[256];
    float* p = out + (size_t)blockIdx.x * DIMK;

    float4 v = ((const float4*)p)[threadIdx.x];
    float ss = v.x * v.x + v.y * v.y + v.z * v.z + v.w * v.w;
    red[threadIdx.x] = ss;
    __syncthreads();
#pragma unroll
    for (int s = 128; s > 0; s >>= 1) {
        if ((int)threadIdx.x < s) red[threadIdx.x] += red[threadIdx.x + s];
        __syncthreads();
    }
    const float inv = rsqrtf(red[0] * (1.f / 1024.f) + 1e-8f);
    const float4 sc = ((const float4*)scale)[threadIdx.x];
    v.x *= inv * sc.x; v.y *= inv * sc.y; v.z *= inv * sc.z; v.w *= inv * sc.w;
    ((float4*)p)[threadIdx.x] = v;
}

extern "C" void kernel_launch(void* const* d_in, const int* in_sizes, int n_in,
                              void* d_out, int out_size, void* d_ws, size_t ws_size,
                              hipStream_t stream) {
    const float* x          = (const float*)d_in[0]; // [4,4096,1024]
    const float* W          = (const float*)d_in[1]; // [1024,4096]
    const float* b          = (const float*)d_in[2]; // [4096]
    const float* input_bias = (const float*)d_in[3]; // [1024]
    const float* scale      = (const float*)d_in[4]; // [1024]
    float* out = (float*)d_out;                      // [4,4096,1024]

    // Workspace layout (floats)
    float* pre  = (float*)d_ws;                                  // 67,108,864
    const size_t NSUM = (size_t)NBATCH * CH * DIMK;              // 262,144
    float* sumF = pre + (size_t)MROWS * NCOLS;
    float* sumC = sumF + NSUM;
    float* sumN = sumC + NSUM;
    float* sumM = sumN + NSUM;
    float* stC  = sumM + NSUM;
    float* stN  = stC + NSUM;
    float* stM  = stN + NSUM;

    dim3 gblock(256);
    dim3 ggrid(NCOLS / TILE_N, MROWS / TILE_M);      // (64, 256)
    gemm_pre_kernel<<<ggrid, gblock, 0, stream>>>(x, W, b, pre);

    const int nChunkThreads = NBATCH * CH * DIMK;    // 262,144
    scan_chunk_summary_kernel<<<nChunkThreads / 256, 256, 0, stream>>>(
        pre, input_bias, sumF, sumC, sumN, sumM);
    scan_compose_kernel<<<(NBATCH * DIMK) / 256, 256, 0, stream>>>(
        sumF, sumC, sumN, sumM, stC, stN, stM);
    scan_replay_kernel<<<nChunkThreads / 256, 256, 0, stream>>>(
        pre, input_bias, stC, stN, stM, out);

    rmsnorm_kernel<<<MROWS, 256, 0, stream>>>(out, scale);
}
